// GCNModel_16045997818346
// MI455X (gfx1250) — matrix operand order repaired and checked by reading, not compiled, not run on hardware
//
#include <hip/hip_runtime.h>

#define FDIM 128
#define CDIM 10

typedef __bf16 bf16;
typedef __attribute__((ext_vector_type(16))) __bf16 v16bf;
typedef __attribute__((ext_vector_type(8)))  float  v8f;

// ---------------------------------------------------------------- utilities
__global__ void fill_f32(float* __restrict__ p, float v, long long n) {
  long long i = (long long)blockIdx.x * blockDim.x + threadIdx.x;
  if (i < n) p[i] = v;
}

__global__ void fill_i32(int* __restrict__ p, int v, int n) {
  int i = blockIdx.x * blockDim.x + threadIdx.x;
  if (i < n) p[i] = v;
}

// in-degree counts (self loop handled as +1 later)
__global__ void count_in(const int* __restrict__ dst, int* __restrict__ cnt, int E) {
  int e = blockIdx.x * blockDim.x + threadIdx.x;
  if (e < E) atomicAdd(&cnt[dst[e]], 1);
}

__global__ void dinv_from_cnt(const int* __restrict__ cnt, float* __restrict__ dinv, int n) {
  int i = blockIdx.x * blockDim.x + threadIdx.x;
  if (i < n) dinv[i] = rsqrtf((float)cnt[i] + 1.0f);
}

// ---------------------------------------------------------------- CSR build (scan)
__global__ __launch_bounds__(256) void scan_partial(
    const int* __restrict__ cnt, int* __restrict__ pre, int* __restrict__ bsums, int n) {
  __shared__ int tmp[256];
  int i = blockIdx.x * 256 + threadIdx.x;
  int v = (i < n) ? cnt[i] : 0;
  tmp[threadIdx.x] = v;
  __syncthreads();
#pragma unroll
  for (int off = 1; off < 256; off <<= 1) {
    int t = (threadIdx.x >= off) ? tmp[threadIdx.x - off] : 0;
    __syncthreads();
    tmp[threadIdx.x] += t;
    __syncthreads();
  }
  if (i < n) pre[i] = tmp[threadIdx.x] - v;         // exclusive within block
  if (threadIdx.x == 255) bsums[blockIdx.x] = tmp[255];
}

// single-block exclusive scan over block sums
__global__ __launch_bounds__(256) void scan_bsums(int* __restrict__ bsums, int nb) {
  __shared__ int tmp[256];
  __shared__ int carry_s;
  if (threadIdx.x == 0) carry_s = 0;
  __syncthreads();
  for (int base = 0; base < nb; base += 256) {
    int i = base + threadIdx.x;
    int v = (i < nb) ? bsums[i] : 0;
    tmp[threadIdx.x] = v;
    __syncthreads();
#pragma unroll
    for (int off = 1; off < 256; off <<= 1) {
      int t = (threadIdx.x >= off) ? tmp[threadIdx.x - off] : 0;
      __syncthreads();
      tmp[threadIdx.x] += t;
      __syncthreads();
    }
    if (i < nb) bsums[i] = tmp[threadIdx.x] - v + carry_s;  // exclusive + carry
    __syncthreads();
    if (threadIdx.x == 0) carry_s += tmp[255];
    __syncthreads();
  }
}

__global__ __launch_bounds__(256) void scan_finalize(
    const int* __restrict__ pre, const int* __restrict__ bsums,
    int* __restrict__ rowptr, int* __restrict__ cursor, int n, int E) {
  int i = blockIdx.x * 256 + threadIdx.x;
  if (i < n) {
    int r = pre[i] + bsums[blockIdx.x];
    rowptr[i] = r;
    cursor[i] = r;
  }
  if (i == 0) rowptr[n] = E;
}

// bucket edges by destination; store src index + normalized edge weight
__global__ void fill_edges(const int* __restrict__ src, const int* __restrict__ dst,
                           const float* __restrict__ dinv, int* __restrict__ cursor,
                           int* __restrict__ esrc, float* __restrict__ ew, int E) {
  int e = blockIdx.x * blockDim.x + threadIdx.x;
  if (e >= E) return;
  int s = src[e], d = dst[e];
  int pos = atomicAdd(&cursor[d], 1);
  esrc[pos] = s;
  ew[pos] = dinv[s] * dinv[d];
}

// ---------------------------------------------------------------- GEMM
// Out[n x 128] = A[n x 128] @ W[128 x 128]
// bf16 WMMA with hi/lo error compensation (~f32 accuracy).
__global__ __launch_bounds__(256) void gemm128_wmma(
    const float* __restrict__ A, const float* __restrict__ W,
    float* __restrict__ Out, int nrows) {
  // W transposed into LDS: column n stored as 8 chunks of 16 K-values,
  // chunk c placed at slot (c ^ (n & 7)) -> bank-conflict-free v16bf loads.
  __shared__ bf16 Whi[FDIM * FDIM];   // 32 KB
  __shared__ bf16 Wlo[FDIM * FDIM];   // 32 KB

  const int tid = threadIdx.x;
#pragma unroll
  for (int it = 0; it < (FDIM * FDIM) / 256; ++it) {
    int idx = tid + it * 256;          // idx = k*128 + n (coalesced read of W)
    int k = idx >> 7, n = idx & 127;
    float w = W[idx];
    bf16 hi = (bf16)w;
    bf16 lo = (bf16)(w - (float)hi);
    int slot = (k >> 4) ^ (n & 7);
    int a = n * FDIM + (slot << 4) + (k & 15);
    Whi[a] = hi;
    Wlo[a] = lo;
  }
  __syncthreads();

  const int wv   = tid >> 5;
  const int lane = tid & 31;
  const int half = lane >> 4;
  const int l16  = lane & 15;

  long long row  = (long long)blockIdx.x * 128 + wv * 16 + l16;
  long long rowc = row < (long long)(nrows - 1) ? row : (long long)(nrows - 1); // clamp: keep EXEC full
  const float* __restrict__ arow = A + rowc * FDIM;

  v8f acc[8] = {};

#pragma unroll
  for (int kk = 0; kk < 4; ++kk) {
    const int k0 = kk * 32;
    float af[16];
    const float4 p0 = *(const float4*)(arow + k0 + half * 16);
    const float4 p1 = *(const float4*)(arow + k0 + half * 16 + 4);
    const float4 p2 = *(const float4*)(arow + k0 + half * 16 + 8);
    const float4 p3 = *(const float4*)(arow + k0 + half * 16 + 12);
    af[0]=p0.x;  af[1]=p0.y;  af[2]=p0.z;  af[3]=p0.w;
    af[4]=p1.x;  af[5]=p1.y;  af[6]=p1.z;  af[7]=p1.w;
    af[8]=p2.x;  af[9]=p2.y;  af[10]=p2.z; af[11]=p2.w;
    af[12]=p3.x; af[13]=p3.y; af[14]=p3.z; af[15]=p3.w;

    v16bf ah, al;
#pragma unroll
    for (int i = 0; i < 16; ++i) {
      bf16 h = (bf16)af[i];
      ah[i] = h;
      al[i] = (bf16)(af[i] - (float)h);
    }

#pragma unroll
    for (int nt = 0; nt < 8; ++nt) {
      const int n    = nt * 16 + l16;
      const int slot = ((kk * 2) + half) ^ (n & 7);
      const v16bf bh = *(const v16bf*)&Whi[n * FDIM + (slot << 4)];
      const v16bf bl = *(const v16bf*)&Wlo[n * FDIM + (slot << 4)];
      acc[nt] = __builtin_amdgcn_wmma_f32_16x16x32_bf16(false, ah, false, bh, (short)0, acc[nt], false, false);
      acc[nt] = __builtin_amdgcn_wmma_f32_16x16x32_bf16(false, al, false, bh, (short)0, acc[nt], false, false);
      acc[nt] = __builtin_amdgcn_wmma_f32_16x16x32_bf16(false, ah, false, bl, (short)0, acc[nt], false, false);
    }
  }

  long long rbase = (long long)blockIdx.x * 128 + wv * 16 + half * 8;
#pragma unroll
  for (int nt = 0; nt < 8; ++nt) {
    int col = nt * 16 + l16;
#pragma unroll
    for (int r = 0; r < 8; ++r) {
      long long rr = rbase + r;
      if (rr < nrows) Out[rr * FDIM + col] = acc[nt][r];
    }
  }
}

// ---------------------------------------------------------------- aggregation (gather, atomic-free)
// One wave per destination node:
//   out[i] = relu?( dinv[i]^2 * hW[i] + bias + sum_e w_e * hW[esrc[e]] )
// Edge metadata is loaded cooperatively (32 records per coalesced load) and
// broadcast via __shfl, so the per-edge feature-row loads have no memory
// dependence between iterations and can be pipelined by the scheduler.
__global__ __launch_bounds__(256) void gather_nodes(
    const float* __restrict__ hw, const int* __restrict__ rowptr,
    const int* __restrict__ esrc, const float* __restrict__ ew,
    const float* __restrict__ dinv, const float* __restrict__ bias,
    float* __restrict__ out, int relu, int n) {
  int i = (blockIdx.x * blockDim.x + threadIdx.x) >> 5;
  int lane = threadIdx.x & 31;
  if (i >= n) return;

  float di = dinv[i];
  const float4 hself = *(const float4*)&hw[(long long)i * FDIM + lane * 4];
  const float4 b     = *(const float4*)&bias[lane * 4];
  float w0 = di * di;
  float4 acc;
  acc.x = w0 * hself.x + b.x;
  acc.y = w0 * hself.y + b.y;
  acc.z = w0 * hself.z + b.z;
  acc.w = w0 * hself.w + b.w;

  int beg = rowptr[i], end = rowptr[i + 1];
  for (int base = beg; base < end; base += 32) {
    int cnt = end - base;
    if (cnt > 32) cnt = 32;
    int   sv = 0;
    float wv = 0.0f;
    if (base + lane < end) {        // coalesced 32-wide metadata load
      sv = esrc[base + lane];
      wv = ew[base + lane];
    }
    for (int j = 0; j < cnt; ++j) {
      int   s = __shfl(sv, j, 32);  // broadcast edge record j to all lanes
      float w = __shfl(wv, j, 32);
      const float4 v = *(const float4*)&hw[(long long)s * FDIM + lane * 4];
      acc.x += w * v.x;
      acc.y += w * v.y;
      acc.z += w * v.z;
      acc.w += w * v.w;
    }
  }
  if (relu) {
    acc.x = fmaxf(acc.x, 0.f); acc.y = fmaxf(acc.y, 0.f);
    acc.z = fmaxf(acc.z, 0.f); acc.w = fmaxf(acc.w, 0.f);
  }
  *(float4*)&out[(long long)i * FDIM + lane * 4] = acc;
}

// ---------------------------------------------------------------- pooling + head
__global__ __launch_bounds__(256) void pool_accum(
    const float* __restrict__ h, const int* __restrict__ batch,
    float* __restrict__ sums, float* __restrict__ cnt, int n) {
  int i = (blockIdx.x * blockDim.x + threadIdx.x) >> 5;
  int lane = threadIdx.x & 31;
  if (i >= n) return;
  int g = batch[i];
  const float4 v = *(const float4*)&h[(long long)i * FDIM + lane * 4];
  float* s = &sums[(long long)g * FDIM + lane * 4];
  atomicAdd(s + 0, v.x);
  atomicAdd(s + 1, v.y);
  atomicAdd(s + 2, v.z);
  atomicAdd(s + 3, v.w);
  if (lane == 0) atomicAdd(&cnt[g], 1.0f);
}

__global__ void pool_linear(const float* __restrict__ sums, const float* __restrict__ cnt,
                            const float* __restrict__ Wlin, const float* __restrict__ blin,
                            float* __restrict__ out, int G) {
  int idx = blockIdx.x * blockDim.x + threadIdx.x;
  if (idx >= G * CDIM) return;
  int g = idx / CDIM, c = idx % CDIM;
  float inv = 1.0f / fmaxf(cnt[g], 1.0f);
  float acc = blin[c];
  for (int k = 0; k < FDIM; ++k)
    acc += sums[g * FDIM + k] * inv * Wlin[k * CDIM + c];
  out[idx] = acc;
}

// ---------------------------------------------------------------- driver
extern "C" void kernel_launch(void* const* d_in, const int* in_sizes, int n_in,
                              void* d_out, int out_size, void* d_ws, size_t ws_size,
                              hipStream_t stream) {
  (void)n_in; (void)ws_size;
  const float* x     = (const float*)d_in[0];
  const int*   ei    = (const int*)d_in[1];
  const int*   batch = (const int*)d_in[2];
  const float* W1 = (const float*)d_in[4];  const float* b1 = (const float*)d_in[5];
  const float* W2 = (const float*)d_in[6];  const float* b2 = (const float*)d_in[7];
  const float* W3 = (const float*)d_in[8];  const float* b3 = (const float*)d_in[9];
  const float* Wl = (const float*)d_in[10]; const float* bl = (const float*)d_in[11];

  const int N = in_sizes[0] / FDIM;
  const int E = in_sizes[1] / 2;
  const int G = out_size / CDIM;
  const int nb = (N + 255) / 256;     // scan blocks
  const int* src = ei;
  const int* dst = ei + E;

  // ---- workspace carve (all 4-byte types) ----
  char* ws = (char*)d_ws;
  size_t ofs = 0;
  auto carve = [&](size_t bytes) { void* p = ws + ofs; ofs += (bytes + 255) & ~(size_t)255; return p; };
  float* bufA   = (float*)carve((size_t)N * FDIM * 4);   // h @ W
  float* bufB   = (float*)carve((size_t)N * FDIM * 4);   // aggregated h
  float* dinv   = (float*)carve((size_t)N * 4);
  float* ew     = (float*)carve((size_t)E * 4);          // edge weights (CSR order)
  float* sums   = (float*)carve((size_t)G * FDIM * 4);
  float* gcnt   = (float*)carve((size_t)G * 4);
  int*   cnt    = (int*)carve((size_t)N * 4);
  int*   pre    = (int*)carve((size_t)N * 4);
  int*   bsums  = (int*)carve((size_t)nb * 4);
  int*   rowptr = (int*)carve((size_t)(N + 1) * 4);
  int*   cursor = (int*)carve((size_t)N * 4);
  int*   esrc   = (int*)carve((size_t)E * 4);            // CSR src indices

  const int thr = 256;
  const int nBlkN  = (N + thr - 1) / thr;
  const int nBlkE  = (E + thr - 1) / thr;
  const int gemmBlocks = (N + 127) / 128;
  const int waveBlocks = (N + 7) / 8;   // one wave (32 lanes) per node

  // ---- build normalized CSR (once; reused by all 3 layers) ----
  fill_i32<<<nBlkN, thr, 0, stream>>>(cnt, 0, N);
  count_in<<<nBlkE, thr, 0, stream>>>(dst, cnt, E);
  dinv_from_cnt<<<nBlkN, thr, 0, stream>>>(cnt, dinv, N);
  scan_partial<<<nb, thr, 0, stream>>>(cnt, pre, bsums, N);
  scan_bsums<<<1, thr, 0, stream>>>(bsums, nb);
  scan_finalize<<<nb, thr, 0, stream>>>(pre, bsums, rowptr, cursor, N, E);
  fill_edges<<<nBlkE, thr, 0, stream>>>(src, dst, dinv, cursor, esrc, ew, E);

  // ---- 3 GCN layers: WMMA GEMM + atomic-free gather (fused bias/self-loop/relu) ----
  auto layer = [&](const float* hin, const float* W, const float* b, int relu) {
    gemm128_wmma<<<gemmBlocks, thr, 0, stream>>>(hin, W, bufA, N);
    gather_nodes<<<waveBlocks, thr, 0, stream>>>(bufA, rowptr, esrc, ew, dinv, b, bufB, relu, N);
  };
  layer(x,    W1, b1, 1);
  layer(bufB, W2, b2, 1);
  layer(bufB, W3, b3, 0);

  // ---- global mean pool + linear head ----
  fill_f32<<<((G * FDIM) + thr - 1) / thr, thr, 0, stream>>>(sums, 0.0f, (long long)G * FDIM);
  fill_f32<<<(G + thr - 1) / thr, thr, 0, stream>>>(gcnt, 0.0f, G);
  pool_accum<<<waveBlocks, thr, 0, stream>>>(bufB, batch, sums, gcnt, N);
  pool_linear<<<(G * CDIM + thr - 1) / thr, thr, 0, stream>>>(sums, gcnt, Wl, bl, (float*)d_out, G);
}
